// HMLSTMCell2_6657199309451
// MI455X (gfx1250) — compile-verified, matrix operand order checked by reading, not run on hardware
//
#include <hip/hip_runtime.h>

typedef __attribute__((ext_vector_type(16))) _Float16 v16h;
typedef __attribute__((ext_vector_type(8)))  float    v8f;

#define HDIM 128
#define GW 513
#define KSTEPS 12          // 384 / 32
#define NTILES_TOT 32      // 512 / 16
#define FRAG_DW 8          // dwords per lane per B fragment
#define PB_DWORDS (KSTEPS * NTILES_TOT * 32 * FRAG_DW)   // 98304 dwords = 384 KB

// ---------------------------------------------------------------------------
// Pre-pack [W;R;U] (384x512 f32) into f16 WMMA-B fragments.
// Fragment f = ks*32 + nt ; lane holds col n = nt*16 + (lane&15),
// K base kb = ks*32 + (lane>=16 ? 8 : 0); halves 0..7 = K kb..kb+7,
// halves 8..15 = K kb+16..kb+23 (matches 16-bit operand VGPR layout).
// ---------------------------------------------------------------------------
__global__ void prepack_weights(const float* __restrict__ W,
                                const float* __restrict__ R,
                                const float* __restrict__ U,
                                unsigned int* __restrict__ PB) {
  int f    = blockIdx.x;            // 0..383
  int ks   = f >> 5;
  int nt   = f & 31;
  int lane = threadIdx.x;           // 0..31
  int col  = nt * 16 + (lane & 15);
  int kb   = ks * 32 + ((lane >= 16) ? 8 : 0);

  unsigned int* dst = PB + ((size_t)f * 32 + lane) * FRAG_DW;
#pragma unroll
  for (int p = 0; p < 8; ++p) {
    union { _Float16 h[2]; unsigned int u; } cv;
#pragma unroll
    for (int q = 0; q < 2; ++q) {
      int K = kb + ((p < 4) ? (2 * p + q) : (16 + 2 * (p - 4) + q));
      float w;
      if (K < 128)      w = W[(size_t)K * GW + col];
      else if (K < 256) w = R[(size_t)(K - 128) * GW + col];
      else              w = U[(size_t)(K - 256) * GW + col];
      cv.h[q] = (_Float16)w;
    }
    dst[p] = cv.u;
  }
}

// Pack column 512 of [W;R;U] (the sz gate) as f32 in the per-lane access order:
// wcol[ks*32 + half*16 + j] , j<8 -> K=kb+j , j>=8 -> K=kb+8+j  (kb=ks*32+half*8)
__global__ void prepack_zcol(const float* __restrict__ W,
                             const float* __restrict__ R,
                             const float* __restrict__ U,
                             float* __restrict__ wcol) {
  int t   = threadIdx.x;            // 0..383
  int ks  = t >> 5;
  int rem = t & 31;
  int h8  = rem >> 4;
  int j   = rem & 15;
  int kb  = ks * 32 + h8 * 8;
  int K   = kb + ((j < 8) ? j : (8 + j));
  float w;
  if (K < 128)      w = W[(size_t)K * GW + 512];
  else if (K < 256) w = R[(size_t)(K - 128) * GW + 512];
  else              w = U[(size_t)(K - 256) * GW + 512];
  wcol[t] = w;
}

// ---------------------------------------------------------------------------
// Main fused kernel: 16-row tile per block, 4 waves, wave w computes gate w
// (128 cols = 8 WMMA N-tiles) over K=384 in 12 steps of 32.
// ---------------------------------------------------------------------------
__global__ __launch_bounds__(128)
void hmlstm_main(const float* __restrict__ hb, const float* __restrict__ hh,
                 const float* __restrict__ ht, const float* __restrict__ c,
                 const float* __restrict__ z,  const float* __restrict__ zb,
                 const float* __restrict__ bias,
                 const unsigned int* __restrict__ PB,
                 const float* __restrict__ wcol,
                 float* __restrict__ out_h, float* __restrict__ out_c,
                 float* __restrict__ out_z) {
  __shared__ float slds[4][16][132];   // stride 132 dwords: lane16 offset = 8*132 ≡ 32 mod 64 banks
  __shared__ float szs[16];

  const int tid   = threadIdx.x;
  const int lane  = tid & 31;
  const int wv    = tid >> 5;                   // gate id 0..3
  const int m0    = blockIdx.x * 16;
  const int mrow  = m0 + (lane & 15);
  const int hhalf = (lane >= 16) ? 1 : 0;

  const float zv_l  = z[mrow];
  const float zbv_l = zb[mrow];

  v8f acc[8] = {};
  float szacc = 0.0f;

  for (int ks = 0; ks < KSTEPS; ++ks) {
    const float* src;
    float mask;
    if (ks < 4)      { src = hb; mask = zbv_l; }
    else if (ks < 8) { src = hh; mask = 1.0f;  }
    else             { src = ht; mask = zv_l;  }

    const int kb = (ks & 3) * 32 + hhalf * 8;
    const float* p = src + (size_t)mrow * HDIM + kb;
    float4 x0 = *(const float4*)(p);
    float4 x1 = *(const float4*)(p + 4);
    float4 x2 = *(const float4*)(p + 16);
    float4 x3 = *(const float4*)(p + 20);
    float xs[16] = {x0.x, x0.y, x0.z, x0.w, x1.x, x1.y, x1.z, x1.w,
                    x2.x, x2.y, x2.z, x2.w, x3.x, x3.y, x3.z, x3.w};

    const float* wc = wcol + ks * 32 + hhalf * 16;
    union { v16h v; _Float16 e[16]; } a;
#pragma unroll
    for (int j = 0; j < 16; ++j) {
      float xm = xs[j] * mask;
      a.e[j] = (_Float16)xm;
      szacc += xm * wc[j];               // f32 partial for the sz column
    }

    // 8 B fragments for this wave's gate; consecutive fragments 1 KB apart
    const v16h* bfr = (const v16h*)(PB +
        ((size_t)(ks * NTILES_TOT + wv * 8) * 32 + lane) * FRAG_DW);
#pragma unroll
    for (int nt = 0; nt < 8; ++nt) {
      v16h bf = bfr[(size_t)nt * 32];
      acc[nt] = __builtin_amdgcn_wmma_f32_16x16x32_f16(
          false, a.v, false, bf, (short)0, acc[nt], false, false);
    }
  }

  // sz: lane l holds half of row (lane&15); fold halves, wave0 publishes
  float szsum = szacc + __shfl_xor(szacc, 16, 32);
  if (wv == 0 && lane < 16) szs[lane] = szsum;

  // dump this wave's 16x128 gate slab (C layout: vgpr r -> M=r(+8), lane&15 -> N)
#pragma unroll
  for (int nt = 0; nt < 8; ++nt) {
    int col = nt * 16 + (lane & 15);
#pragma unroll
    for (int r = 0; r < 8; ++r) {
      slds[wv][r + hhalf * 8][col] = acc[nt][r];
    }
  }
  __syncthreads();

  // epilogue: thread t owns column j = t for all 16 rows
  const int j = tid;
  const float bi = bias[j];
  const float bg = bias[HDIM + j];
  const float bo = bias[2 * HDIM + j];
  const float bff = bias[3 * HDIM + j];
#pragma unroll 4
  for (int r = 0; r < 16; ++r) {
    int m = m0 + r;
    float si = slds[0][r][j] + bi;
    float sg = slds[1][r][j] + bg;
    float so = slds[2][r][j] + bo;
    float sf = slds[3][r][j] + bff;
    float iv = 1.0f / (1.0f + __expf(-si));
    float gv = tanhf(sg);
    float ov = 1.0f / (1.0f + __expf(-so));
    float fv = 1.0f / (1.0f + __expf(-sf));
    float ig = iv * gv;
    float zr  = z[m];
    float zbr = zb[m];
    float cv  = c[(size_t)m * HDIM + j];
    float cn  = (zr == 1.0f) ? ig : ((zbr == 0.0f) ? cv : cv * fv + ig);
    float hn  = ((zr == 0.0f) && (zbr == 0.0f)) ? hh[(size_t)m * HDIM + j]
                                                : tanhf(cn) * ov;
    out_h[(size_t)m * HDIM + j] = hn;
    out_c[(size_t)m * HDIM + j] = cn;
  }
  if (tid < 16) {
    // forward of round_st(hard_sigm(sz)) with a=1, th=0.5  <=>  sz > 0
    float sz = szs[tid] + bias[512];
    out_z[m0 + tid] = (sz > 0.0f) ? 1.0f : 0.0f;
  }
}

// ---------------------------------------------------------------------------
extern "C" void kernel_launch(void* const* d_in, const int* in_sizes, int n_in,
                              void* d_out, int out_size, void* d_ws, size_t ws_size,
                              hipStream_t stream) {
  const float* hb = (const float*)d_in[0];
  const float* hh = (const float*)d_in[1];
  const float* ht = (const float*)d_in[2];
  const float* c  = (const float*)d_in[3];
  const float* z  = (const float*)d_in[4];
  const float* zb = (const float*)d_in[5];
  const float* W  = (const float*)d_in[6];
  const float* R  = (const float*)d_in[7];
  const float* U  = (const float*)d_in[8];
  const float* b  = (const float*)d_in[9];

  unsigned int* PB = (unsigned int*)d_ws;
  float* wcol = (float*)((char*)d_ws + (size_t)PB_DWORDS * sizeof(unsigned int));

  const int Bn = in_sizes[1] / HDIM;          // rows

  prepack_weights<<<KSTEPS * NTILES_TOT, 32, 0, stream>>>(W, R, U, PB);
  prepack_zcol<<<1, KSTEPS * 32, 0, stream>>>(W, R, U, wcol);

  float* out_h = (float*)d_out;
  float* out_c = out_h + (size_t)Bn * HDIM;
  float* out_z = out_c + (size_t)Bn * HDIM;

  hmlstm_main<<<Bn / 16, 128, 0, stream>>>(hb, hh, ht, c, z, zb, b, PB, wcol,
                                           out_h, out_c, out_z);
}
